// MultiHeadAttention_64364379898038
// MI455X (gfx1250) — compile-verified, matrix operand order checked by reading, not run on hardware
//
#include <hip/hip_runtime.h>

typedef __attribute__((ext_vector_type(16))) _Float16 v16h;
typedef __attribute__((ext_vector_type(8)))  _Float16 v8h;
typedef __attribute__((ext_vector_type(8)))  float    v8f;
typedef __attribute__((ext_vector_type(4)))  _Float16 v4h;

#define DEV __device__ __forceinline__

// ---------------------------------------------------------------------------
// Fragment loader for V_WMMA_F32_16X16X32_F16 (wave32).
// 16-bit A-matrix 16x32 layout (ISA 7.12.2): lane L -> row = L%16, group g=L/16;
// per lane the fragment is two CONTIGUOUS 16B chunks:
//   elements 0..7  = K 8g..8g+7      at base + row*stride + 8g
//   elements 8..15 = K 16+8g..16+8g+7 at base + row*stride + 8g + 16
// -> two b128 loads (global or DS) + a register-level concat. B fragments use
// the same shape with "row" = N column.
// ---------------------------------------------------------------------------
DEV v16h load_frag_row(const _Float16* __restrict__ base, int stride, int lane) {
    int row = lane & 15, g = lane >> 4;
    const _Float16* p = base + (unsigned)(row * stride + 8 * g);
    v8h lo = *(const v8h*)p;          // 16B chunk, K = 8g..8g+7
    v8h hi = *(const v8h*)(p + 16);   // 16B chunk, K = 16+8g..16+8g+7
    return __builtin_shufflevector(lo, hi, 0, 1, 2, 3, 4, 5, 6, 7,
                                           8, 9, 10, 11, 12, 13, 14, 15);
}

DEV v8f wmma_f16(v16h a, v16h b, v8f c) {
    return __builtin_amdgcn_wmma_f32_16x16x32_f16(
        /*neg_a=*/false, a, /*neg_b=*/false, b,
        /*c_mod=*/(short)0, c, /*reuse_a=*/false, /*reuse_b=*/false);
}

// CDNA5 async global->LDS copy (16B per lane) + split-counter waits.
DEV void async_copy_b128(unsigned lds_off, const _Float16* gptr) {
    unsigned long long ga = (unsigned long long)(uintptr_t)gptr;
    asm volatile("global_load_async_to_lds_b128 %0, %1, off"
                 :: "v"(lds_off), "v"(ga) : "memory");
}
DEV void wait_asynccnt0() { asm volatile("s_wait_asynccnt 0" ::: "memory"); }
DEV void wait_dscnt0()    { asm volatile("s_wait_dscnt 0"    ::: "memory"); }

// ---------------------------------------------------------------------------
// f32 -> f16 elementwise convert (float4 per thread)
// ---------------------------------------------------------------------------
__global__ __launch_bounds__(256)
void f32_to_f16_kernel(const float4* __restrict__ src, _Float16* __restrict__ dst, int n4) {
    int i = blockIdx.x * 256 + threadIdx.x;
    if (i < n4) {
        float4 x = src[i];
        v4h h; h[0] = (_Float16)x.x; h[1] = (_Float16)x.y;
               h[2] = (_Float16)x.z; h[3] = (_Float16)x.w;
        *(v4h*)(dst + (size_t)i * 4) = h;
    }
}

// ---------------------------------------------------------------------------
// W[K,N] f32 -> Wt[N,K] f16 (32x32 LDS tile for coalescing both sides)
// ---------------------------------------------------------------------------
__global__ __launch_bounds__(1024)
void transpose_f32_to_f16_kernel(const float* __restrict__ W, _Float16* __restrict__ Wt,
                                 int N, int K) {
    __shared__ _Float16 tile[32][33];
    int n0 = blockIdx.x * 32, k0 = blockIdx.y * 32;
    int tx = threadIdx.x, ty = threadIdx.y;
    tile[ty][tx] = (_Float16)W[(size_t)(k0 + ty) * N + n0 + tx];
    __syncthreads();
    Wt[(size_t)(n0 + ty) * K + k0 + tx] = tile[tx][ty];
}

// ---------------------------------------------------------------------------
// Per-head V transpose:  Vp[b,t,h*dh+d]  ->  VpT[(b*H+h)*dh+d, t]
// so P@V B-fragments read consecutive keys (b128 loads).
// ---------------------------------------------------------------------------
__global__ __launch_bounds__(1024)
void transpose_v_kernel(const _Float16* __restrict__ Vp, _Float16* __restrict__ VpT,
                        int T, int D, int dh, int H) {
    __shared__ _Float16 tile[32][33];
    int t0 = blockIdx.x * 32;          // token tile
    int d0 = blockIdx.y * 32;          // dh tile
    int bh = blockIdx.z;               // b*H + h
    int b = bh / H, h = bh - b * H;
    int tx = threadIdx.x, ty = threadIdx.y;
    tile[ty][tx] = Vp[((size_t)b * T + t0 + ty) * D + h * dh + d0 + tx];
    __syncthreads();
    VpT[((size_t)bh * dh + d0 + ty) * T + t0 + tx] = tile[tx][ty];
}

// ---------------------------------------------------------------------------
// GEMM:  Y[M,N] = (X[M,K] @ Wt[N,K]^T + bias[N]) * oscale   (all-f16 operands)
// 256 threads = 8 waves; block tile 128x128; K step 32.
// Tiles staged to LDS via async global->LDS b128 copies.
// ---------------------------------------------------------------------------
template <typename Tout>
__global__ __launch_bounds__(256)
void gemm_f16_kernel(const _Float16* __restrict__ X, const _Float16* __restrict__ Wt,
                     const float* __restrict__ bias, Tout* __restrict__ Y,
                     int M, int N, int K, float oscale) {
    __shared__ __align__(16) _Float16 lds_a[128 * 32];   // [m][k]
    __shared__ __align__(16) _Float16 lds_b[128 * 32];   // [n][k]
    const int tid  = threadIdx.x;
    const int wave = tid >> 5, lane = tid & 31;
    const int m0 = blockIdx.x * 128, n0 = blockIdx.y * 128;

    const unsigned lds_a_base = (unsigned)(uintptr_t)lds_a;
    const unsigned lds_b_base = (unsigned)(uintptr_t)lds_b;

    v8f acc[8] = {};

    for (int k0 = 0; k0 < K; k0 += 32) {
        wait_dscnt0();          // previous iteration's LDS fragment reads done
        __syncthreads();
        // stage A and B tiles: 128 rows x 32 f16 = 512 x 16B chunks each
#pragma unroll
        for (int i = 0; i < 2; ++i) {
            int c  = tid + i * 256;        // chunk id 0..511
            int r  = c >> 2;               // tile row
            int cc = (c & 3) * 8;          // f16 col offset within row
            async_copy_b128(lds_a_base + (unsigned)c * 16,
                            X + (size_t)(m0 + r) * K + k0 + cc);
            async_copy_b128(lds_b_base + (unsigned)c * 16,
                            Wt + (size_t)(n0 + r) * K + k0 + cc);
        }
        wait_asynccnt0();
        __syncthreads();

        v16h af = load_frag_row(lds_a + wave * 16 * 32, 32, lane);
#pragma unroll
        for (int t = 0; t < 8; ++t) {
            v16h bf = load_frag_row(lds_b + t * 16 * 32, 32, lane);
            acc[t] = wmma_f16(af, bf, acc[t]);
        }
    }

    const int n = lane & 15, g = lane >> 4;
    Tout* yb = Y + (size_t)m0 * N + n0;
#pragma unroll
    for (int t = 0; t < 8; ++t) {
        float bv = bias[n0 + t * 16 + n];
#pragma unroll
        for (int r = 0; r < 8; ++r) {
            int m = wave * 16 + r + 8 * g;
            yb[(unsigned)(m * N + t * 16 + n)] = (Tout)((acc[t][r] + bv) * oscale);
        }
    }
}

// ---------------------------------------------------------------------------
// Flash attention: one wave per (b, h, 16-row q tile). dh = 64, causal.
// Softmax scale pre-folded into Q. Row-sum via "ones-column" WMMA;
// row-max via batched cross-lane reductions. V read from per-head
// transposed layout so all fragments use b128 consecutive-k loads.
// ---------------------------------------------------------------------------
__global__ __launch_bounds__(32)
void flash_attn_kernel(const _Float16* __restrict__ Qp, const _Float16* __restrict__ Kp,
                       const _Float16* __restrict__ VpT, _Float16* __restrict__ Op,
                       int T, int D, int dh, int H) {
    __shared__ __align__(16) _Float16 lds_p[16 * 32];  // P block, A-fragment order

    const int lane = threadIdx.x;
    const int q0 = blockIdx.x * 16;
    const int h  = blockIdx.y;
    const int b  = blockIdx.z;
    const size_t baseBD = (size_t)b * T * D + (size_t)h * dh;
    const _Float16* vthead = VpT + (size_t)(b * H + h) * dh * T;

    const _Float16* qbase = Qp + baseBD + (size_t)q0 * D;
    const v16h qf0 = load_frag_row(qbase,      D, lane);   // kdim 0..31
    const v16h qf1 = load_frag_row(qbase + 32, D, lane);   // kdim 32..63

    v16h ones;
#pragma unroll
    for (int e = 0; e < 16; ++e) ones[e] = (_Float16)1.0f;

    v8f o[4] = {};
    v8f lacc = {};                         // running row-sum (replicated per lane)
    float mrow[8];
#pragma unroll
    for (int r = 0; r < 8; ++r) mrow[r] = -__builtin_inff();

    const int n = lane & 15, g = lane >> 4;

    for (int j0 = 0; j0 <= q0 + 15; j0 += 32) {
        // S = Q @ K^T for 32 keys, as two 16x16 f32 blocks
        const _Float16* kb0 = Kp + baseBD + (size_t)j0 * D;
        const _Float16* kb1 = kb0 + (size_t)16 * D;
        v8f s0 = {}, s1 = {};
        s0 = wmma_f16(qf0, load_frag_row(kb0,      D, lane), s0);
        s0 = wmma_f16(qf1, load_frag_row(kb0 + 32, D, lane), s0);
        s1 = wmma_f16(qf0, load_frag_row(kb1,      D, lane), s1);
        s1 = wmma_f16(qf1, load_frag_row(kb1 + 32, D, lane), s1);

        // causal mask (row r+8g lives in register r, lane half g)
        float sv0[8], sv1[8], x[8];
#pragma unroll
        for (int r = 0; r < 8; ++r) {
            int qrow = q0 + r + 8 * g;
            sv0[r] = (j0 + n      <= qrow) ? s0[r] : -__builtin_inff();
            sv1[r] = (j0 + 16 + n <= qrow) ? s1[r] : -__builtin_inff();
            x[r]   = fmaxf(sv0[r], sv1[r]);
        }
        // batched row-max across the 16-lane half
#pragma unroll
        for (int mask = 8; mask >= 1; mask >>= 1) {
            float t[8];
#pragma unroll
            for (int r = 0; r < 8; ++r) t[r] = __shfl_xor(x[r], mask, 32);
#pragma unroll
            for (int r = 0; r < 8; ++r) x[r] = fmaxf(x[r], t[r]);
        }
#pragma unroll
        for (int r = 0; r < 8; ++r) {
            float rm    = fmaxf(x[r], mrow[r]);
            float alpha = __expf(mrow[r] - rm);
            mrow[r] = rm;
            lacc[r] *= alpha;
            o[0][r] *= alpha; o[1][r] *= alpha; o[2][r] *= alpha; o[3][r] *= alpha;
            float p0 = __expf(sv0[r] - rm);
            float p1 = __expf(sv1[r] - rm);
            int m = r + 8 * g;
            lds_p[m * 32 + n]      = (_Float16)p0;
            lds_p[m * 32 + 16 + n] = (_Float16)p1;
        }
        wait_dscnt0();   // wave-local DS stores visible before DS fragment reads

        v16h pf = load_frag_row(lds_p, 32, lane);           // P as A fragment
        lacc = wmma_f16(pf, ones, lacc);                    // row-sum accumulate
        const _Float16* vtb = vthead + j0;                  // [dh][T] + key offset
#pragma unroll
        for (int d = 0; d < 4; ++d) {
            v16h vf = load_frag_row(vtb + (size_t)(d * 16) * T, T, lane);
            o[d] = wmma_f16(pf, vf, o[d]);
        }
    }

    _Float16* obase = Op + baseBD + (size_t)q0 * D;
#pragma unroll
    for (int d = 0; d < 4; ++d) {
#pragma unroll
        for (int r = 0; r < 8; ++r) {
            int m = r + 8 * g;
            obase[(unsigned)(m * D + d * 16 + n)] = (_Float16)(o[d][r] / lacc[r]);
        }
    }
}

// ---------------------------------------------------------------------------
extern "C" void kernel_launch(void* const* d_in, const int* in_sizes, int n_in,
                              void* d_out, int out_size, void* d_ws, size_t ws_size,
                              hipStream_t stream) {
    (void)in_sizes; (void)n_in; (void)out_size; (void)ws_size;

    const float* q  = (const float*)d_in[0];
    const float* k  = (const float*)d_in[1];
    const float* v  = (const float*)d_in[2];
    const float* Wq = (const float*)d_in[3];
    const float* bq = (const float*)d_in[4];
    const float* Wk = (const float*)d_in[5];
    const float* bk = (const float*)d_in[6];
    const float* Wv = (const float*)d_in[7];
    const float* bv = (const float*)d_in[8];
    const float* Wo = (const float*)d_in[9];
    const float* bo = (const float*)d_in[10];
    float* out = (float*)d_out;

    const int B = 4, T = 2048, D = 1024, H = 16, dh = 64;
    const int M = B * T;                                   // 8192 rows
    const float qscale = 0.125f;                           // 1/sqrt(dh)

    const size_t actBytes = (size_t)M * D * sizeof(_Float16);   // 16 MB
    const size_t wBytes   = (size_t)D * D * sizeof(_Float16);   // 2 MB
    char* ws = (char*)d_ws;
    _Float16* qh  = (_Float16*)(ws);                       // f16 copies of inputs
    _Float16* kh  = (_Float16*)(ws + 1 * actBytes);
    _Float16* vh  = (_Float16*)(ws + 2 * actBytes);
    _Float16* Qp  = (_Float16*)(ws + 3 * actBytes);        // projected activations
    _Float16* Kp  = (_Float16*)(ws + 4 * actBytes);
    _Float16* Vp  = (_Float16*)(ws + 5 * actBytes);
    _Float16* Oa  = (_Float16*)(ws + 6 * actBytes);
    _Float16* VpT = (_Float16*)(ws + 7 * actBytes);        // per-head V^T
    _Float16* WqT = (_Float16*)(ws + 8 * actBytes);        // transposed f16 weights
    _Float16* WkT = (_Float16*)(ws + 8 * actBytes + 1 * wBytes);
    _Float16* WvT = (_Float16*)(ws + 8 * actBytes + 2 * wBytes);
    _Float16* WoT = (_Float16*)(ws + 8 * actBytes + 3 * wBytes);

    // 1) precision/layout conversion (L2-resident, one pass)
    const int n4 = (M * D) / 4;
    f32_to_f16_kernel<<<n4 / 256, 256, 0, stream>>>((const float4*)q, qh, n4);
    f32_to_f16_kernel<<<n4 / 256, 256, 0, stream>>>((const float4*)k, kh, n4);
    f32_to_f16_kernel<<<n4 / 256, 256, 0, stream>>>((const float4*)v, vh, n4);
    dim3 tgrid(D / 32, D / 32), tblk(32, 32);
    transpose_f32_to_f16_kernel<<<tgrid, tblk, 0, stream>>>(Wq, WqT, D, D);
    transpose_f32_to_f16_kernel<<<tgrid, tblk, 0, stream>>>(Wk, WkT, D, D);
    transpose_f32_to_f16_kernel<<<tgrid, tblk, 0, stream>>>(Wv, WvT, D, D);
    transpose_f32_to_f16_kernel<<<tgrid, tblk, 0, stream>>>(Wo, WoT, D, D);

    // 2) Q/K/V projections (async-LDS staged f16 WMMA GEMM); softmax scale
    //    folded into the Q projection epilogue.
    dim3 ggrid(M / 128, D / 128), gblk(256);
    gemm_f16_kernel<_Float16><<<ggrid, gblk, 0, stream>>>(qh, WqT, bq, Qp, M, D, D, qscale);
    gemm_f16_kernel<_Float16><<<ggrid, gblk, 0, stream>>>(kh, WkT, bk, Kp, M, D, D, 1.0f);
    gemm_f16_kernel<_Float16><<<ggrid, gblk, 0, stream>>>(vh, WvT, bv, Vp, M, D, D, 1.0f);

    // 2b) per-head V transpose for consecutive-key fragment loads
    dim3 vgrid(T / 32, dh / 32, B * H), vblk(32, 32);
    transpose_v_kernel<<<vgrid, vblk, 0, stream>>>(Vp, VpT, T, D, dh, H);

    // 3) causal flash attention (16 q-rows per wave)
    dim3 agrid(T / 16, H, B);
    flash_attn_kernel<<<agrid, 32, 0, stream>>>(Qp, Kp, VpT, Oa, T, D, dh, H);

    // 4) output projection
    gemm_f16_kernel<float><<<ggrid, gblk, 0, stream>>>(Oa, WoT, bo, out, M, D, D, 1.0f);
}
